// RPN_12549894439514
// MI455X (gfx1250) — matrix operand order repaired and checked by reading, not compiled, not run on hardware
//
#include <hip/hip_runtime.h>
#include <hip/hip_bf16.h>

// ---------------- types ----------------
typedef __attribute__((ext_vector_type(16))) __bf16 v16bf;
typedef __attribute__((ext_vector_type(8)))  __bf16 v8bf;
typedef __attribute__((ext_vector_type(8)))  float  v8f;
typedef __attribute__((ext_vector_type(8)))  unsigned short v8us;

// problem constants
#define BATCH   8
#define CIN     512
#define HH      50
#define WW      64
#define HWPIX   (HH*WW)          // 3200
#define NPIX    (BATCH*HWPIX)    // 25600
#define COUT    512
#define KTOT    (9*CIN)          // 4608
#define KSTEP   32
#define KITERS  (KTOT/KSTEP)     // 144
#define NANCH   9
#define KBOX    (HWPIX*NANCH)    // 28800 per image
#define PRENMS  6000
#define CAP     6144
#define POSTN   300
#define IMGW    1024.0f
#define IMGH    800.0f

// workspace layout (bytes)
#define WS_XPACK  0u                 // bf16 [NPIX][512] NHWC       26,214,400
#define WS_WPACK  26214400u          // bf16 [9][512co][512ci]       4,718,592
#define WS_HPACK  30932992u          // bf16 [64][512]                  65,536
#define WS_FEAT   30998528u          // bf16 [NPIX][512]            26,214,400
#define WS_HEAD   57212928u          // f32  [NPIX][64]              6,553,600
#define WS_BOXES  63766528u          // f32  [8][28800][4]           3,686,400
#define WS_SCORES 67452928u          // f32  [8][28800]                921,600

// NMS dynamic-LDS layout (bytes)
#define L_SBOX   0u                  // CAP*16 = 98304
#define L_SSC    98304u              // CAP*4  = 24576 -> 122880
#define L_BINS   122880u             // 1024  -> 123904
#define L_SV     123904u             // 1024  -> 124928
#define L_SI     124928u             // 1024  -> 125952
#define L_MISC   125952u             // 64    -> 126016
#define NMS_LDS  126016u

__device__ __forceinline__ unsigned short f2bf(float f) {
    unsigned int u = __builtin_bit_cast(unsigned int, f);
    u += 0x7FFFu + ((u >> 16) & 1u);            // RNE
    return (unsigned short)(u >> 16);
}

__device__ __forceinline__ v16bf cat16(v8bf lo, v8bf hi) {
    return __builtin_shufflevector(lo, hi, 0,1,2,3,4,5,6,7,8,9,10,11,12,13,14,15);
}

// --- CDNA5 async global->LDS copy (ASYNCcnt path) ---
__device__ __forceinline__ void async_b128(void* lds, const void* g) {
    unsigned loff = (unsigned)(unsigned long long)lds;
    asm volatile("global_load_async_to_lds_b128 %0, %1, off"
                 :: "v"(loff), "v"(g) : "memory");
}
__device__ __forceinline__ void async_wait0() {
    asm volatile("s_wait_asynccnt 0" ::: "memory");
}

// ================= pack kernels (one-shot bf16 repacks) =================
__global__ void __launch_bounds__(256)
pack_x(const float* __restrict__ x, unsigned short* __restrict__ xp) {
    __shared__ unsigned short tile[64][65];        // [ci][x], padded
    const int t   = threadIdx.x;
    const int ny  = blockIdx.x;                    // n*50 + y
    const int n   = ny / HH, y = ny - n * HH;
    const int ci0 = blockIdx.y * 64;
    #pragma unroll
    for (int r = (t >> 6); r < 64; r += 4) {       // read coalesced along x
        float v = x[(((size_t)n * CIN + ci0 + r) * HH + y) * WW + (t & 63)];
        tile[r][t & 63] = f2bf(v);
    }
    __syncthreads();
    #pragma unroll
    for (int xc = (t >> 6); xc < 64; xc += 4) {    // write coalesced along ci
        xp[((size_t)ny * WW + xc) * CIN + ci0 + (t & 63)] = tile[t & 63][xc];
    }
}

__global__ void __launch_bounds__(256)
pack_w(const float* __restrict__ w1, unsigned short* __restrict__ wp) {
    const int t = blockIdx.x * 256 + threadIdx.x;
    if (t >= 9 * COUT * CIN) return;
    const int ci   = t & 511;
    const int co   = (t >> 9) & 511;
    const int ktap = t >> 18;
    wp[t] = f2bf(w1[((size_t)co * CIN + ci) * 9 + ktap]);
}

__global__ void __launch_bounds__(256)
pack_h(const float* __restrict__ wcls, const float* __restrict__ wbox,
       unsigned short* __restrict__ hp) {
    const int t = blockIdx.x * 256 + threadIdx.x;
    if (t >= 64 * CIN) return;
    const int ci = t & 511, col = t >> 9;
    float v = 0.0f;
    if (col < 18)      v = wcls[(size_t)col * CIN + ci];
    else if (col < 54) v = wbox[(size_t)(col - 18) * CIN + ci];
    hp[t] = f2bf(v);
}

// ================= Kernel 1: 3x3 conv, bf16 WMMA, async double-buffered =================
__global__ void __launch_bounds__(256)
conv3x3_wmma(const unsigned short* __restrict__ xpack,
             const unsigned short* __restrict__ wpack,
             const float* __restrict__ b1, unsigned short* __restrict__ feat) {
    __shared__ __align__(16) unsigned short Abuf[2][128 * KSTEP];   // 2 x 8 KB
    __shared__ __align__(16) unsigned short Bbuf[2][128 * KSTEP];   // 2 x 8 KB

    const int tid  = threadIdx.x;
    const int lane = tid & 31;
    const int wv   = tid >> 5;
    const int pix0 = blockIdx.x * 128;
    const int co0  = blockIdx.y * 128;

    const int pA   = tid & 127;
    const int g0   = tid >> 7;
    const int pixA = pix0 + pA;
    const int nA   = pixA / HWPIX;
    const int remA = pixA - nA * HWPIX;
    const int yA   = remA >> 6;
    const int xA   = remA & 63;

    v8f acc[8] = {};

    auto stage = [&](int buf, int k0) {
        const int ktap = k0 >> 9;
        const int ci0  = k0 & 511;
        const int ky   = ktap / 3, kx = ktap - 3 * ky;
        const int yy   = yA + ky - 1, xx = xA + kx - 1;
        if ((unsigned)yy < (unsigned)HH && (unsigned)xx < (unsigned)WW) {
            const unsigned short* src =
                &xpack[((size_t)((nA * HH + yy) * WW + xx)) * CIN + ci0];
            async_b128(&Abuf[buf][pA * KSTEP + g0 * 8],       src + g0 * 8);
            async_b128(&Abuf[buf][pA * KSTEP + (g0 + 2) * 8], src + (g0 + 2) * 8);
        } else {
            v8us z = {};
            *(v8us*)&Abuf[buf][pA * KSTEP + g0 * 8]       = z;
            *(v8us*)&Abuf[buf][pA * KSTEP + (g0 + 2) * 8] = z;
        }
        const unsigned short* wsrc =
            &wpack[((size_t)ktap * COUT + (co0 + pA)) * CIN + ci0];
        async_b128(&Bbuf[buf][pA * KSTEP + g0 * 8],       wsrc + g0 * 8);
        async_b128(&Bbuf[buf][pA * KSTEP + (g0 + 2) * 8], wsrc + (g0 + 2) * 8);
    };

    auto compute = [&](int buf) {
        const int row = wv * 16 + (lane & 15);
        const int kb  = (lane < 16) ? 0 : 8;
        v16bf a = cat16(*(const v8bf*)&Abuf[buf][row * KSTEP + kb],
                        *(const v8bf*)&Abuf[buf][row * KSTEP + kb + 16]);
        const int kb2 = (lane < 16) ? 0 : 8;
        v16bf bfr[8];
        #pragma unroll
        for (int j = 0; j < 8; ++j) {
            const int colb = j * 16 + (lane & 15);
            bfr[j] = cat16(*(const v8bf*)&Bbuf[buf][colb * KSTEP + kb2],
                           *(const v8bf*)&Bbuf[buf][colb * KSTEP + kb2 + 16]);
        }
        #pragma unroll
        for (int j = 0; j < 8; ++j)
            acc[j] = __builtin_amdgcn_wmma_f32_16x16x32_bf16(
                false, a, false, bfr[j], (short)0, acc[j], false, false);
    };

    stage(0, 0);
    async_wait0();
    __syncthreads();
    for (int it = 0; it < KITERS; ++it) {
        const int cur = it & 1;
        if (it + 1 < KITERS) stage(cur ^ 1, (it + 1) * KSTEP);
        compute(cur);
        async_wait0();
        __syncthreads();
    }

    const int mh = (lane >> 4) * 8;
    #pragma unroll
    for (int j = 0; j < 8; ++j) {
        const int co = co0 + j * 16 + (lane & 15);
        const float bias = b1[co];
        #pragma unroll
        for (int v = 0; v < 8; ++v) {
            const int pixel = pix0 + wv * 16 + v + mh;
            float val = acc[j][v] + bias;
            val = val > 0.0f ? val : 0.0f;
            feat[(size_t)pixel * CIN + co] = f2bf(val);
        }
    }
}

// ================= Kernel 2: 1x1 heads GEMM — whole B resident in LDS =================
// dynamic shared: 64 cols x 512 ci bf16 = 65536 B, staged once via async
__global__ void __launch_bounds__(256)
head_wmma(const unsigned short* __restrict__ feat,
          const unsigned short* __restrict__ hpack,
          const float* __restrict__ bcls, const float* __restrict__ bbox,
          float* __restrict__ head) {
    extern __shared__ __align__(16) unsigned short Bh[];   // [64][512]

    const int tid  = threadIdx.x;
    const int lane = tid & 31;
    const int wv   = tid >> 5;
    const int pix0 = blockIdx.x * 128;

    #pragma unroll
    for (int c = 0; c < 16; ++c) {
        const int chunk = tid + c * 256;                   // 0..4095, 16B each
        async_b128(&Bh[chunk * 8], &hpack[chunk * 8]);
    }
    v8f acc[4] = {};
    async_wait0();
    __syncthreads();

    #pragma unroll 4
    for (int k0 = 0; k0 < CIN; k0 += KSTEP) {
        const int row = pix0 + wv * 16 + (lane & 15);
        const int kb  = (lane < 16) ? 0 : 8;
        const unsigned short* ap = &feat[(size_t)row * CIN + k0 + kb];
        v16bf a = cat16(*(const v8bf*)(ap), *(const v8bf*)(ap + 16));

        const int kb2 = (lane < 16) ? 0 : 8;
        v16bf bfr[4];
        #pragma unroll
        for (int j = 0; j < 4; ++j) {
            const int colb = j * 16 + (lane & 15);
            bfr[j] = cat16(*(const v8bf*)&Bh[colb * CIN + k0 + kb2],
                           *(const v8bf*)&Bh[colb * CIN + k0 + kb2 + 16]);
        }
        #pragma unroll
        for (int j = 0; j < 4; ++j)
            acc[j] = __builtin_amdgcn_wmma_f32_16x16x32_bf16(
                false, a, false, bfr[j], (short)0, acc[j], false, false);
    }

    const int mh = (lane >> 4) * 8;
    #pragma unroll
    for (int j = 0; j < 4; ++j) {
        const int c = j * 16 + (lane & 15);
        float bias = 0.0f;
        if (c < 18)      bias = bcls[c];
        else if (c < 54) bias = bbox[c - 18];
        #pragma unroll
        for (int v = 0; v < 8; ++v) {
            const int pixel = pix0 + wv * 16 + v + mh;
            head[(size_t)pixel * 64 + c] = acc[j][v] + bias;
        }
    }
}

// ================= Kernel 3: score + box decode =================
__global__ void __launch_bounds__(256)
decode_kernel(const float* __restrict__ head, float* __restrict__ boxes,
              float* __restrict__ scores) {
    const int t = blockIdx.x * blockDim.x + threadIdx.x;
    if (t >= BATCH * KBOX) return;
    const int n  = t / KBOX;
    const int r  = t - n * KBOX;
    const int a  = r % NANCH;
    const int pw = r / NANCH;
    const int y  = pw >> 6;
    const int x  = pw & 63;
    const int pixel = n * HWPIX + pw;
    const float* h = &head[(size_t)pixel * 64];

    const int ri = a / 3, si = a - 3 * ri;
    const float ratio = (ri == 0) ? 0.5f : (ri == 1 ? 1.0f : 2.0f);
    const float scale = (si == 0) ? 8.0f : (si == 1 ? 16.0f : 32.0f);
    const float sr = __builtin_sqrtf(ratio);
    const float ah = 16.0f * scale * sr;
    const float aw = 16.0f * scale / sr;
    const float acx = x * 16.0f + 8.0f;
    const float acy = y * 16.0f + 8.0f;

    const float s0 = h[2 * a], s1 = h[2 * a + 1];
    const float fg = 1.0f / (1.0f + __expf(s0 - s1));

    const float d0 = h[18 + 4 * a + 0], d1 = h[18 + 4 * a + 1];
    const float d2 = h[18 + 4 * a + 2], d3 = h[18 + 4 * a + 3];
    const float cx = d0 * aw + acx;
    const float cy = d1 * ah + acy;
    const float pw2 = __expf(d2) * aw;
    const float ph2 = __expf(d3) * ah;

    float x1 = fminf(fmaxf(cx - 0.5f * pw2, 0.0f), IMGW);
    float y1 = fminf(fmaxf(cy - 0.5f * ph2, 0.0f), IMGH);
    float x2 = fminf(fmaxf(cx + 0.5f * pw2, 0.0f), IMGW);
    float y2 = fminf(fmaxf(cy + 0.5f * ph2, 0.0f), IMGH);

    float* b = &boxes[(size_t)t * 4];
    b[0] = x1; b[1] = y1; b[2] = x2; b[3] = y2;
    const bool ok = ((x2 - x1) >= 16.0f) & ((y2 - y1) >= 16.0f);
    scores[t] = ok ? fg : -__builtin_inff();
}

// ================= Kernel 4: pre-NMS selection + LDS-resident NMS =================
// one block per image; dynamic shared NMS_LDS bytes
__global__ void __launch_bounds__(256)
nms_kernel(const float* __restrict__ boxes, const float* __restrict__ scores,
           float* __restrict__ out) {
    extern __shared__ __align__(16) char nsmem[];
    float* sbox = (float*)(nsmem + L_SBOX);     // [CAP][4]
    float* ssc  = (float*)(nsmem + L_SSC);      // [CAP]
    int*   bins = (int*)(nsmem + L_BINS);       // [256]
    float* sv   = (float*)(nsmem + L_SV);       // [256]
    int*   si   = (int*)(nsmem + L_SI);         // [256]
    int*   cnt  = (int*)(nsmem + L_MISC);       // [0]=count, [1]=threshold bin
    float* sb   = (float*)(nsmem + L_MISC + 8); // selected box

    const int n   = blockIdx.x;
    const int tid = threadIdx.x;
    const float* gs = scores + (size_t)n * KBOX;
    const float* gb = boxes  + (size_t)n * KBOX * 4;

    // --- histogram of valid scores (sigmoid -> (0,1)) ---
    bins[tid] = 0;
    __syncthreads();
    for (int j = tid; j < KBOX; j += 256) {
        float v = gs[j];
        if (v >= 0.0f) {
            int b = (int)(v * 256.0f);
            b = b > 255 ? 255 : b;
            atomicAdd(&bins[b], 1);
        }
    }
    __syncthreads();
    if (tid == 0) {
        int cum = 0, bt = 0;
        for (int b = 255; b >= 0; --b) {
            cum += bins[b];
            if (cum >= PRENMS) { bt = b; break; }
        }
        cnt[1] = bt;
    }
    __syncthreads();
    const float thr = cnt[1] * (1.0f / 256.0f);

    // --- deterministic compaction (prefix-sum, ordered) into LDS ---
    int mycnt = 0;
    for (int j = tid; j < KBOX; j += 256)
        if (gs[j] >= thr) mycnt++;
    si[tid] = mycnt;
    __syncthreads();
    if (tid == 0) {
        int run = 0;
        for (int t = 0; t < 256; ++t) { int c = si[t]; si[t] = run; run += c; }
        cnt[0] = run;
    }
    __syncthreads();
    int pos = si[tid];
    for (int j = tid; j < KBOX; j += 256) {
        float v = gs[j];
        if (v >= thr) {
            if (pos < CAP) {
                ssc[pos] = v;
                sbox[pos * 4 + 0] = gb[j * 4 + 0];
                sbox[pos * 4 + 1] = gb[j * 4 + 1];
                sbox[pos * 4 + 2] = gb[j * 4 + 2];
                sbox[pos * 4 + 3] = gb[j * 4 + 3];
            }
            pos++;
        }
    }
    __syncthreads();
    int count = cnt[0] > CAP ? CAP : cnt[0];
    if (count == 0) {                       // degenerate guard
        if (tid == 0) {
            ssc[0] = -__builtin_inff();
            sbox[0] = sbox[1] = sbox[2] = sbox[3] = 0.0f;
            cnt[0] = 1;
        }
        __syncthreads();
        count = 1;
    }

    // --- 300 iterations of argmax + IoU suppression, all in LDS ---
    for (int it = 0; it < POSTN; ++it) {
        float best = -__builtin_inff(); int bi = 0;
        for (int j = tid; j < count; j += 256) {
            float v = ssc[j];
            if (v > best) { best = v; bi = j; }
        }
        sv[tid] = best; si[tid] = bi;
        __syncthreads();
        for (int st = 128; st > 0; st >>= 1) {
            if (tid < st) {
                float vo = sv[tid + st]; int io = si[tid + st];
                if (vo > sv[tid] || (vo == sv[tid] && io < si[tid])) {
                    sv[tid] = vo; si[tid] = io;
                }
            }
            __syncthreads();
        }
        if (tid == 0) {
            const int b = si[0];
            sb[0] = sbox[b * 4 + 0]; sb[1] = sbox[b * 4 + 1];
            sb[2] = sbox[b * 4 + 2]; sb[3] = sbox[b * 4 + 3];
            ssc[b] = -__builtin_inff();
            float* ro = &out[((size_t)n * POSTN + it) * 4];
            ro[0] = sb[0]; ro[1] = sb[1]; ro[2] = sb[2]; ro[3] = sb[3];
            out[(size_t)BATCH * POSTN * 4 + (size_t)n * POSTN + it] = (float)n;
        }
        __syncthreads();
        const float X1 = sb[0], Y1 = sb[1], X2 = sb[2], Y2 = sb[3];
        const float a1 = (X2 - X1) * (Y2 - Y1);
        for (int j = tid; j < count; j += 256) {
            const float bx1 = sbox[j * 4 + 0], by1 = sbox[j * 4 + 1];
            const float bx2 = sbox[j * 4 + 2], by2 = sbox[j * 4 + 3];
            const float lx = fmaxf(X1, bx1), ly = fmaxf(Y1, by1);
            const float rx = fminf(X2, bx2), ry = fminf(Y2, by2);
            const float iw = fmaxf(rx - lx, 0.0f), ih = fmaxf(ry - ly, 0.0f);
            const float inter = iw * ih;
            const float a2 = (bx2 - bx1) * (by2 - by1);
            const float iou = inter / (a1 + a2 - inter + 1e-9f);
            if (iou >= 0.7f) ssc[j] = -__builtin_inff();
        }
        __syncthreads();
    }
}

extern "C" void kernel_launch(void* const* d_in, const int* in_sizes, int n_in,
                              void* d_out, int out_size, void* d_ws, size_t ws_size,
                              hipStream_t stream) {
    (void)in_sizes; (void)n_in; (void)out_size; (void)ws_size;
    const float* x      = (const float*)d_in[0];
    const float* w1     = (const float*)d_in[1];
    const float* b1     = (const float*)d_in[2];
    const float* w_cls  = (const float*)d_in[3];
    const float* b_cls  = (const float*)d_in[4];
    const float* w_bbox = (const float*)d_in[5];
    const float* b_bbox = (const float*)d_in[6];

    unsigned short* xpack = (unsigned short*)((char*)d_ws + WS_XPACK);
    unsigned short* wpack = (unsigned short*)((char*)d_ws + WS_WPACK);
    unsigned short* hpack = (unsigned short*)((char*)d_ws + WS_HPACK);
    unsigned short* feat  = (unsigned short*)((char*)d_ws + WS_FEAT);
    float* head   = (float*)((char*)d_ws + WS_HEAD);
    float* boxes  = (float*)((char*)d_ws + WS_BOXES);
    float* scores = (float*)((char*)d_ws + WS_SCORES);
    float* out    = (float*)d_out;

    pack_x<<<dim3(BATCH * HH, CIN / 64), 256, 0, stream>>>(x, xpack);
    pack_w<<<(9 * COUT * CIN + 255) / 256, 256, 0, stream>>>(w1, wpack);
    pack_h<<<(64 * CIN + 255) / 256, 256, 0, stream>>>(w_cls, w_bbox, hpack);

    conv3x3_wmma<<<dim3(NPIX / 128, COUT / 128), 256, 0, stream>>>(xpack, wpack, b1, feat);
    head_wmma<<<NPIX / 128, 256, 65536, stream>>>(feat, hpack, b_cls, b_bbox, head);
    decode_kernel<<<(BATCH * KBOX + 255) / 256, 256, 0, stream>>>(head, boxes, scores);
    nms_kernel<<<BATCH, 256, NMS_LDS, stream>>>(boxes, scores, out);
}
